// GCN_Hetro_30846455120312
// MI455X (gfx1250) — matrix operand order repaired and checked by reading, not compiled, not run on hardware
//
#include <hip/hip_runtime.h>
#include <hip/hip_bf16.h>
#include <cstdint>
#include <cstddef>

#define N_NODESC 50000
#define N_EDGESC 400000
#define N_GRAPHSC 64
#define EMBC 64
#define HFC 128
#define EPSC 1e-5f
#define ROWT 5            // 16-row tiles per block -> 80 rows, 50000/80 = 625 blocks

typedef __attribute__((ext_vector_type(16))) _Float16 v16h;
typedef __attribute__((ext_vector_type(8)))  _Float16 v8h;
typedef __attribute__((ext_vector_type(8)))  float    v8f;

// ---------------------------------------------------------------------------
// Pack 24 weight matrices (4 layers x 3 relations x {self,neigh}) into the
// WMMA B-fragment lane order: element p = ((tile_n*4 + kk)*32 + lane)*16 + j
// holds W[k][col], k = kk*32 + (lane>>4)*16 + j, col = tile_n*16 + (lane&15).
// Layer-1 matrices (K=64) are zero-padded to K=128.
// ---------------------------------------------------------------------------
__global__ void k_pack(const float* __restrict__ W1s, const float* __restrict__ W1n,
                       const float* __restrict__ Ws,  const float* __restrict__ Wn,
                       _Float16* __restrict__ wpack) {
    int t = blockIdx.x * blockDim.x + threadIdx.x;
    if (t >= 24 * 16384) return;
    int mat = t >> 14;
    int p = t & 16383;
    int j = p & 15, lane = (p >> 4) & 31, kk = (p >> 9) & 3, tn = (p >> 11) & 7;
    int k = kk * 32 + (lane >> 4) * 16 + j;
    int col = tn * 16 + (lane & 15);
    int which = mat & 1, li = mat >> 1, l = li / 3, i = li % 3;
    float v;
    if (l == 0) {
        const float* W = (which ? W1n : W1s) + i * EMBC * HFC;
        v = (k < EMBC) ? W[k * HFC + col] : 0.f;
    } else {
        const float* W = (which ? Wn : Ws) + (size_t)((l - 1) * 3 + i) * HFC * HFC;
        v = W[k * HFC + col];
    }
    wpack[t] = (_Float16)v;
}

// Embedding gather -> f16 features, cols [64,128) zero-padded.
__global__ void k_embed(const int* __restrict__ hI, const int* __restrict__ pI,
                        const int* __restrict__ hpI,
                        const float* __restrict__ eh, const float* __restrict__ ep,
                        const float* __restrict__ ehp,
                        _Float16* __restrict__ x16) {
    int t = blockIdx.x * blockDim.x + threadIdx.x;
    if (t >= 3 * N_NODESC * 32) return;
    int i = t / (N_NODESC * 32);
    int r = t % (N_NODESC * 32);
    int n = r >> 5;
    int c0 = (r & 31) * 4;
    const int* idx = (i == 0) ? hI : (i == 1) ? pI : hpI;
    const float* emb = (i == 0) ? eh : (i == 1) ? ep : ehp;
    int row = idx[n];
    _Float16* o = x16 + ((size_t)i * N_NODESC + n) * HFC + c0;
#pragma unroll
    for (int q = 0; q < 4; q++) {
        int c = c0 + q;
        o[q] = (_Float16)((c < EMBC) ? emb[row * EMBC + c] : 0.f);
    }
}

// -------------------------- CSR construction (once) ------------------------
__global__ void k_deg(const int* __restrict__ dst, int* __restrict__ degi) {
    int t = blockIdx.x * blockDim.x + threadIdx.x;
    if (t >= 3 * N_EDGESC) return;
    int i = t / N_EDGESC, e = t % N_EDGESC;
    atomicAdd(&degi[i * N_NODESC + dst[i * N_EDGESC + e]], 1);
}

// One block per relation: blocked exclusive scan of degi -> rowptr.
__global__ void k_rowptr(const int* __restrict__ degi, int* __restrict__ rowptr) {
    int i = blockIdx.x;
    int t = threadIdx.x;
    __shared__ int sh[256];
    __shared__ int sbase;
    if (t == 0) sbase = 0;
    __syncthreads();
    for (int start = 0; start < N_NODESC; start += 256) {
        int idx = start + t;
        int v = (idx < N_NODESC) ? degi[i * N_NODESC + idx] : 0;
        sh[t] = v;
        __syncthreads();
        for (int o = 1; o < 256; o <<= 1) {
            int x = (t >= o) ? sh[t - o] : 0;
            __syncthreads();
            sh[t] += x;
            __syncthreads();
        }
        if (idx < N_NODESC) rowptr[i * (N_NODESC + 1) + idx] = sbase + sh[t] - v;
        int total = sh[255];
        __syncthreads();
        if (t == 0) sbase += total;
        __syncthreads();
    }
    if (t == 0) rowptr[i * (N_NODESC + 1) + N_NODESC] = sbase;   // == N_EDGESC
}

__global__ void k_fill(const int* __restrict__ src, const int* __restrict__ dst,
                       const int* __restrict__ rowptr, int* __restrict__ fill,
                       int* __restrict__ csr) {
    int t = blockIdx.x * blockDim.x + threadIdx.x;
    if (t >= 3 * N_EDGESC) return;
    int i = t / N_EDGESC, e = t % N_EDGESC;
    int d = dst[i * N_EDGESC + e];
    int pos = atomicAdd(&fill[i * N_NODESC + d], 1);
    csr[(size_t)i * N_EDGESC + rowptr[i * (N_NODESC + 1) + d] + pos] =
        src[i * N_EDGESC + e];
}

__global__ void k_cnt(const int* __restrict__ seg, float* __restrict__ cnt) {
    int t = blockIdx.x * blockDim.x + threadIdx.x;
    if (t >= 3 * N_NODESC) return;
    int i = t / N_NODESC, n = t % N_NODESC;
    atomicAdd(&cnt[i * N_GRAPHSC + seg[i * N_NODESC + n]], 1.f);
}

// ------------------- Neighbor mean: atomic-free CSR gather -----------------
// Thread = (relation, node, 8-col chunk). 16 consecutive lanes process the
// same node's edge list in lockstep -> 256B coalesced reads per edge.
__global__ void k_gather(const _Float16* __restrict__ x16, const int* __restrict__ rowptr,
                         const int* __restrict__ csr, _Float16* __restrict__ m16) {
    int t = blockIdx.x * blockDim.x + threadIdx.x;
    if (t >= 3 * N_NODESC * 16) return;
    int ch = t & 15;
    int r = t >> 4;
    int n = r % N_NODESC;
    int i = r / N_NODESC;
    const int* rp = rowptr + i * (N_NODESC + 1);
    int e0 = rp[n], e1 = rp[n + 1];
    const int* cs = csr + (size_t)i * N_EDGESC;
    float acc[8] = {0.f, 0.f, 0.f, 0.f, 0.f, 0.f, 0.f, 0.f};
    for (int e = e0; e < e1; e++) {
        int s = cs[e];
        v8h v = *(const v8h*)(x16 + ((size_t)i * N_NODESC + s) * HFC + ch * 8);
#pragma unroll
        for (int q = 0; q < 8; q++) acc[q] += (float)v[q];
    }
    float inv = (e1 > e0) ? 1.f / (float)(e1 - e0) : 0.f;   // zero-degree -> 0
    v8h o;
#pragma unroll
    for (int q = 0; q < 8; q++) o[q] = (_Float16)(acc[q] * inv);
    *(v8h*)(m16 + ((size_t)i * N_NODESC + n) * HFC + ch * 8) = o;
}

// ---------------------------------------------------------------------------
// WMMA GEMM: out = PReLU(x@Wself + nmean@Wneigh + bias).
// Block = 80 rows x 128 cols; A/M tiles staged in LDS (shared by all 8
// waves); each wave owns one 16-col tile and sweeps 5 row tiles, reusing its
// B fragment -> 40 v_wmma per wave. One-pass BN stats via colsum/colsq.
// ---------------------------------------------------------------------------
__global__ __launch_bounds__(256)
void k_gemm(const _Float16* __restrict__ x16, const _Float16* __restrict__ m16,
            const _Float16* __restrict__ wpack,
            const float* __restrict__ bias_l, const float* __restrict__ a_l, int l,
            _Float16* __restrict__ pre16, float* __restrict__ colsum,
            float* __restrict__ colsq) {
    int i = blockIdx.y;
    int m0 = blockIdx.x * (16 * ROWT);
    int tid = threadIdx.x;
    int wave = tid >> 5;
    int lane = tid & 31;
    int mrow = lane & 15;
    int hh = lane >> 4;

    __shared__ _Float16 shA[16 * ROWT][HFC + 8];   // +8 halves row pad
    __shared__ _Float16 shM[16 * ROWT][HFC + 8];

    // Cooperative stage: 80 rows x 128 halves per buffer, v8h granularity.
    for (int p = tid; p < 16 * ROWT * 16; p += 256) {
        int rr = p >> 4;
        int cc = (p & 15) * 8;
        *(v8h*)&shA[rr][cc] = *(const v8h*)(x16 + ((size_t)i * N_NODESC + m0 + rr) * HFC + cc);
        *(v8h*)&shM[rr][cc] = *(const v8h*)(m16 + ((size_t)i * N_NODESC + m0 + rr) * HFC + cc);
    }
    __syncthreads();

    int matS = (l * 3 + i) * 2;
    const _Float16* Bs = wpack + (size_t)matS * 16384 + wave * 2048 + lane * 16;
    const _Float16* Bn = Bs + 16384;

    v8f acc[ROWT] = {};
#pragma unroll
    for (int kk = 0; kk < 4; kk++) {
        union { v16h v; v8h h[2]; } bs, bn2;
        bs.v  = *(const v16h*)(Bs + kk * 512);
        bn2.v = *(const v16h*)(Bn + kk * 512);
#pragma unroll
        for (int rt = 0; rt < ROWT; rt++) {
            union { v16h v; v8h h[2]; } a;
            a.h[0] = *(const v8h*)&shA[rt * 16 + mrow][kk * 32 + hh * 16];
            a.h[1] = *(const v8h*)&shA[rt * 16 + mrow][kk * 32 + hh * 16 + 8];
            acc[rt] = __builtin_amdgcn_wmma_f32_16x16x32_f16(false, a.v, false, bs.v,
                                                             (short)0, acc[rt], false, false);
            a.h[0] = *(const v8h*)&shM[rt * 16 + mrow][kk * 32 + hh * 16];
            a.h[1] = *(const v8h*)&shM[rt * 16 + mrow][kk * 32 + hh * 16 + 8];
            acc[rt] = __builtin_amdgcn_wmma_f32_16x16x32_f16(false, a.v, false, bn2.v,
                                                             (short)0, acc[rt], false, false);
        }
    }

    // Epilogue: C layout -> VGPR r holds row (r + 8*hh), col = wave*16 + mrow.
    int col = wave * 16 + mrow;
    float bb = bias_l[i * HFC + col];
    float al = a_l[i * HFC + col];
    float s1 = 0.f, s2 = 0.f;
#pragma unroll
    for (int rt = 0; rt < ROWT; rt++) {
        _Float16* P = pre16 + ((size_t)i * N_NODESC + m0 + rt * 16) * HFC + col;
#pragma unroll
        for (int r = 0; r < 8; r++) {
            float v = acc[rt][r] + bb;
            v = v > 0.f ? v : al * v;   // PReLU
            s1 += v;
            s2 += v * v;
            P[(size_t)(r + 8 * hh) * HFC] = (_Float16)v;
        }
    }
    atomicAdd(&colsum[i * HFC + col], s1);
    atomicAdd(&colsq[i * HFC + col], s2);
}

// BatchNorm apply + per-graph readout accumulation into d_out.
__global__ void k_bn(const _Float16* __restrict__ pre16, const float* __restrict__ colsum,
                     const float* __restrict__ colsq,
                     const float* __restrict__ gamma_l, const float* __restrict__ beta_l,
                     const int* __restrict__ seg, int l,
                     _Float16* __restrict__ x16, float* __restrict__ out) {
    int t = blockIdx.x * blockDim.x + threadIdx.x;
    if (t >= 3 * N_NODESC * 32) return;
    int i = t / (N_NODESC * 32);
    int r = t % (N_NODESC * 32);
    int n = r >> 5;
    int c0 = (r & 31) * 4;
    int g = seg[i * N_NODESC + n];
    size_t base = ((size_t)i * N_NODESC + n) * HFC + c0;
    float* og = out + (size_t)g * 1536 + i * 512 + l * 128 + c0;
    const float invN = 1.f / (float)N_NODESC;
#pragma unroll
    for (int q = 0; q < 4; q++) {
        int col = c0 + q;
        float mu = colsum[i * HFC + col] * invN;
        float var = colsq[i * HFC + col] * invN - mu * mu;
        float x = (float)pre16[base + q];
        float y = gamma_l[i * HFC + col] * (x - mu) * rsqrtf(var + EPSC)
                  + beta_l[i * HFC + col];
        x16[base + q] = (_Float16)y;          // next layer input
        atomicAdd(og + q, y);                 // graph readout sum
    }
}

__global__ void k_final(float* __restrict__ out, const float* __restrict__ cnt) {
    int t = blockIdx.x * blockDim.x + threadIdx.x;
    if (t >= N_GRAPHSC * 1536) return;
    int g = t / 1536;
    int q = t % 1536;
    int i = q / 512;
    float c = cnt[i * N_GRAPHSC + g];
    c = c > 1.f ? c : 1.f;
    out[t] /= c;
}

extern "C" void kernel_launch(void* const* d_in, const int* in_sizes, int n_in,
                              void* d_out, int out_size, void* d_ws, size_t ws_size,
                              hipStream_t stream) {
    const int*   h_idx  = (const int*)d_in[0];
    const int*   p_idx  = (const int*)d_in[1];
    const int*   hp_idx = (const int*)d_in[2];
    const int*   src    = (const int*)d_in[3];
    const int*   dst    = (const int*)d_in[4];
    const int*   seg    = (const int*)d_in[5];
    const float* emb_h  = (const float*)d_in[6];
    const float* emb_p  = (const float*)d_in[7];
    const float* emb_hp = (const float*)d_in[8];
    const float* W1s    = (const float*)d_in[9];
    const float* W1n    = (const float*)d_in[10];
    const float* b1     = (const float*)d_in[11];
    const float* a1     = (const float*)d_in[12];
    const float* g1     = (const float*)d_in[13];
    const float* be1    = (const float*)d_in[14];
    const float* Ws     = (const float*)d_in[15];
    const float* Wn     = (const float*)d_in[16];
    const float* b_     = (const float*)d_in[17];
    const float* a_     = (const float*)d_in[18];
    const float* g_     = (const float*)d_in[19];
    const float* be_    = (const float*)d_in[20];

    char* ws = (char*)d_ws;
    size_t off = 0;
    auto alloc = [&](size_t bytes) -> void* {
        void* p = ws + off;
        off += (bytes + 255) & ~(size_t)255;
        return p;
    };
    _Float16* x16    = (_Float16*)alloc((size_t)3 * N_NODESC * HFC * 2);
    _Float16* m16    = (_Float16*)alloc((size_t)3 * N_NODESC * HFC * 2);
    _Float16* pre16  = (_Float16*)alloc((size_t)3 * N_NODESC * HFC * 2);
    int*      degi   = (int*)     alloc((size_t)3 * N_NODESC * 4);
    int*      rowptr = (int*)     alloc((size_t)3 * (N_NODESC + 1) * 4);
    int*      fill   = (int*)     alloc((size_t)3 * N_NODESC * 4);
    int*      csr    = (int*)     alloc((size_t)3 * N_EDGESC * 4);
    float*    cnt    = (float*)   alloc((size_t)3 * N_GRAPHSC * 4);
    float*    colsum = (float*)   alloc((size_t)3 * HFC * 4);
    float*    colsq  = (float*)   alloc((size_t)3 * HFC * 4);
    _Float16* wpack  = (_Float16*)alloc((size_t)24 * 16384 * 2);
    float*    out    = (float*)d_out;

    hipMemsetAsync(d_out, 0, (size_t)N_GRAPHSC * 1536 * 4, stream);
    hipMemsetAsync(degi, 0, (size_t)3 * N_NODESC * 4, stream);
    hipMemsetAsync(fill, 0, (size_t)3 * N_NODESC * 4, stream);
    hipMemsetAsync(cnt, 0, (size_t)3 * N_GRAPHSC * 4, stream);

    // One-time structure: weight pack, embeddings, CSR-by-dst, graph counts.
    k_pack<<<(24 * 16384 + 255) / 256, 256, 0, stream>>>(W1s, W1n, Ws, Wn, wpack);
    k_embed<<<(3 * N_NODESC * 32 + 255) / 256, 256, 0, stream>>>(
        h_idx, p_idx, hp_idx, emb_h, emb_p, emb_hp, x16);
    k_deg<<<(3 * N_EDGESC + 255) / 256, 256, 0, stream>>>(dst, degi);
    k_rowptr<<<3, 256, 0, stream>>>(degi, rowptr);
    k_fill<<<(3 * N_EDGESC + 255) / 256, 256, 0, stream>>>(src, dst, rowptr, fill, csr);
    k_cnt<<<(3 * N_NODESC + 255) / 256, 256, 0, stream>>>(seg, cnt);

    for (int l = 0; l < 4; l++) {
        hipMemsetAsync(colsum, 0, (size_t)3 * HFC * 4, stream);
        hipMemsetAsync(colsq, 0, (size_t)3 * HFC * 4, stream);

        k_gather<<<(3 * N_NODESC * 16 + 255) / 256, 256, 0, stream>>>(
            x16, rowptr, csr, m16);

        const float* bl  = (l == 0) ? b1  : b_  + (size_t)(l - 1) * 3 * HFC;
        const float* al  = (l == 0) ? a1  : a_  + (size_t)(l - 1) * 3 * HFC;
        const float* gl  = (l == 0) ? g1  : g_  + (size_t)(l - 1) * 3 * HFC;
        const float* bel = (l == 0) ? be1 : be_ + (size_t)(l - 1) * 3 * HFC;

        dim3 gdim(N_NODESC / (16 * ROWT), 3);
        k_gemm<<<gdim, 256, 0, stream>>>(x16, m16, wpack, bl, al, l,
                                         pre16, colsum, colsq);
        k_bn<<<(3 * N_NODESC * 32 + 255) / 256, 256, 0, stream>>>(
            pre16, colsum, colsq, gl, bel, seg, l, x16, out);
    }
    k_final<<<(N_GRAPHSC * 1536 + 255) / 256, 256, 0, stream>>>(out, cnt);
}
